// Net_71631464563168
// MI455X (gfx1250) — matrix operand order repaired and checked by reading, not compiled, not run on hardware
//
#include <hip/hip_runtime.h>
#include <hip/hip_bf16.h>
#include <math.h>

// ---------------- types ----------------
typedef __bf16 bf16_t;
typedef bf16_t v16bf __attribute__((ext_vector_type(16)));
typedef bf16_t v8bf  __attribute__((ext_vector_type(8)));
typedef float  v8f   __attribute__((ext_vector_type(8)));

#define HH   640
#define WW   640
#define MH   40
#define MW   40
#define NC   8
#define KPER 400
#define NB   4
// conv tile
#define TW 64
#define TH 16

// ============================================================
// Kernel 1: mask generation (sigmoid -> exact top-k threshold
// via bit-pattern binary search -> binarize -> tile)
// ============================================================
__global__ __launch_bounds__(256) void mask_kernel(
    const float* __restrict__ wmask, const float* __restrict__ my1,
    float* __restrict__ outAdj, float* __restrict__ outBin,
    float* __restrict__ wsMask)
{
    __shared__ float sSig[MH * MW];
    __shared__ int   sCnt;
    const int b   = blockIdx.x;
    const int tid = threadIdx.x;
    const float scale = my1[b];

    // sigmoid(scale * w_eff); ACS columns 18..22 forced to 1e7
    for (int j = tid; j < MH * MW; j += 256) {
        int col  = j % MW;
        float wv = (col >= 18 && col < 23) ? 1.0e7f : wmask[j];
        sSig[j]  = 1.0f / (1.0f + expf(-scale * wv));
    }
    __syncthreads();

    // exact k-th largest: max t (uint bits, all values > 0) with
    // count(x >= t) >= KPER.  Handles the 1e7-column ties.
    unsigned lo = 0u, hi = 0x7f7fffffu;
    while (lo < hi) {
        unsigned mid = lo + ((hi - lo + 1u) >> 1);
        if (tid == 0) sCnt = 0;
        __syncthreads();
        int c = 0;
        for (int j = tid; j < MH * MW; j += 256)
            c += (__float_as_uint(sSig[j]) >= mid) ? 1 : 0;
        atomicAdd(&sCnt, c);
        __syncthreads();
        int cnt = sCnt;
        __syncthreads();
        if (cnt >= KPER) lo = mid; else hi = mid - 1u;
    }
    const float thr = __uint_as_float(lo);

    // binarize: kept (s>=thr, s>0) -> probs=(s+1)/2 > 0.5 -> hard=1
    for (int j = tid; j < MH * MW; j += 256) {
        float hard = (sSig[j] >= thr) ? 1.0f : 0.0f;
        sSig[j] = hard;
        outBin[b * MH * MW + j] = hard;
        wsMask[b * MH * MW + j] = hard;
    }
    __syncthreads();

    // tile 40x40 -> 640x640 (mask_adj output)
    for (int j = tid; j < HH * WW; j += 256) {
        int y = j / WW, x = j - y * WW;
        outAdj[(size_t)b * HH * WW + j] = sSig[(y % MH) * MW + (x % MW)];
    }
}

// ============================================================
// Kernel 2: one SPIRiT step = complex 3x3 conv (8->8 coils) as
// implicit GEMM over 16 "channels" (re0..7, im0..7) with
// v_wmma_f32_16x16x32_bf16, + branchless data-consistency blend.
//   D[16oc x 16px] = sum_{k<144} A[16 x 144] * B[144 x 16px]
//   (K = tap*16 + ch, padded to 160 -> 5 WMMAs per pixel tile)
// ============================================================
__global__ __launch_bounds__(256) void spirit_step(
    const float* __restrict__ src,    // previous pred (or x for stack 0)
    const float* __restrict__ x0,     // original k-space data
    const float* __restrict__ maskv,  // [B][40][40] binary mask (ws)
    const float* __restrict__ wr,     // this stack's real weights [8][8][3][3]
    const float* __restrict__ wi,     // this stack's imag weights
    float* __restrict__ dst,
    int applyMask)                    // 1 for stack 0 (src = raw x)
{
    // [y][x][ch16] bf16; row TH+2 (==18) is a zero pad row so the K=160
    // padding (virtual tap 9) reads zeros with no EXEC divergence.
    __shared__ alignas(32) bf16_t sIn[TH + 3][TW + 2][16];
    // A matrix per tap: [tap][outch][inch]; tap 4 (center) and tap 9
    // (K padding) are zero.
    __shared__ alignas(32) bf16_t sA[10][16][16];
    // per-workgroup mask tile (kills the %40 chains in the hot loop)
    __shared__ float sM[TH][TW];

    const int b   = blockIdx.z;
    const int tX0 = blockIdx.x * TW;
    const int tY0 = blockIdx.y * TH;
    const int tid = threadIdx.x;
    const float* mb = maskv + b * (MH * MW);

    // ---- build A (complex -> real 16x16 block per tap) ----
    for (int idx = tid; idx < 10 * 16 * 16; idx += 256) {
        int t = idx >> 8; int m = (idx >> 4) & 15; int c = idx & 15;
        float v = 0.0f;
        if (t < 9 && t != 4) {
            int o = m & 7, i = c & 7;
            float wrv = wr[(o * 8 + i) * 9 + t];
            float wiv = wi[(o * 8 + i) * 9 + t];
            v = (m < 8) ? ((c < 8) ? wrv : -wiv)
                        : ((c < 8) ? wiv :  wrv);
        }
        sA[t][m][c] = (bf16_t)v;
    }
    // ---- stage mask tile ----
    for (int idx = tid; idx < TH * TW; idx += 256) {
        int r = idx >> 6, cx = idx & 63;
        sM[r][cx] = mb[((tY0 + r) % MH) * MW + ((tX0 + cx) % MW)];
    }
    // ---- zero pad row 18 ----
    for (int idx = tid; idx < (TW + 2) * 16; idx += 256)
        sIn[TH + 2][idx >> 4][idx & 15] = (bf16_t)0.0f;
    // ---- stage input tile + halo, fp32 -> bf16, coalesced per coil ----
    const int NSITE = (TH + 2) * (TW + 2);   // 18*66
    for (int task = tid; task < NSITE * NC; task += 256) {
        int coil = task / NSITE;
        int site = task - coil * NSITE;
        int sy = site / (TW + 2), sx = site - sy * (TW + 2);
        int gy = tY0 + sy - 1,    gx = tX0 + sx - 1;
        float re = 0.0f, im = 0.0f;
        if (gy >= 0 && gy < HH && gx >= 0 && gx < WW) {
            size_t base = ((((size_t)b * NC + coil) * HH + gy) * WW + gx) * 2;
            re = src[base]; im = src[base + 1];
            if (applyMask) {
                float mv = mb[(gy % MH) * MW + (gx % MW)];
                re *= mv; im *= mv;
            }
        }
        sIn[sy][sx][coil]     = (bf16_t)re;
        sIn[sy][sx][coil + 8] = (bf16_t)im;
    }
    __syncthreads();

    const int lane = tid & 31;
    const int wave = tid >> 5;
    const int half = lane >> 4;   // K-block select / re-im select
    const int n    = lane & 15;   // A row (outch) == B col (pixel)

    // ---- per-lane A fragments (this code's contiguous k order:
    //      k = e + 16*half -> elems 0..15 = tap 2c+half ch[0..15]) ----
    v16bf afrag[5];
    #pragma unroll
    for (int c = 0; c < 5; ++c) {
        union { v16bf v; v8bf h[2]; } au;
        au.h[0] = *(const v8bf*)&sA[2 * c + half][n][0];
        au.h[1] = *(const v8bf*)&sA[2 * c + half][n][8];
        afrag[c] = au.v;
    }

    // ---- 8 pixel tiles per wave; tile index = wave + 8*i, so the
    //      column base is fixed and the row advances by 2 per step:
    //      all addressing strength-reduces to pointer increments. ----
    const int r0 = wave >> 2;              // 0 or 1
    const int cb = (wave & 3) << 4;        // 0,16,32,48
    const int gx = tX0 + cb + n;
    const bf16_t* pIn = &sIn[r0][cb + n][0];
    const float*  pM  = &sM[r0][cb + n];
    size_t base = ((((size_t)b * NC) * HH + (tY0 + r0)) * WW + gx) * 2 + half;
    const size_t coilStride = (size_t)HH * WW * 2;

    for (int i = 0; i < 8; ++i) {
        // ---- issue the DC operands FIRST so the 8 global loads are a
        //      single clause whose latency hides under the WMMA chain ----
        float mv = *pM;
        float xv[8];
        #pragma unroll
        for (int vr = 0; vr < 8; ++vr)
            xv[vr] = x0[base + (size_t)vr * coilStride];

        v8f acc = {};
        #pragma unroll
        for (int c = 0; c < 5; ++c) {
            // B uses the same contiguous order: k = e + 16*half -> one full tap per
            // lane-half, 16 contiguous bf16 channels in LDS.
            int tap = 2 * c + half;           // tap 9 -> padded zero row
            int dy  = tap / 3, dx = tap - dy * 3;
            v16bf bfrag = *(const v16bf*)(pIn + (dy * (TW + 2) + dx) * 16);
            acc = __builtin_amdgcn_wmma_f32_16x16x32_bf16(
                      false, afrag[c], false, bfrag, (short)0, acc, false, false);
        }
        // ---- branchless DC: pred = y + m*(x0 - y)  (m binary) ----
        #pragma unroll
        for (int vr = 0; vr < 8; ++vr) {
            // C/D layout: VGPR vr holds outch m = vr + 8*half
            // -> coil = vr, re/im = half
            float y = acc[vr];
            dst[base + (size_t)vr * coilStride] = fmaf(mv, xv[vr] - y, y);
        }
        base += (size_t)2 * WW * 2;        // next row pair
        pIn  += 2 * (TW + 2) * 16;
        pM   += 2 * TW;
    }
}

// ============================================================
extern "C" void kernel_launch(void* const* d_in, const int* in_sizes, int n_in,
                              void* d_out, int out_size, void* d_ws, size_t ws_size,
                              hipStream_t stream) {
    (void)in_sizes; (void)n_in; (void)out_size; (void)ws_size;
    const float* x  = (const float*)d_in[0];   // [4,8,640,640,2]
    const float* m1 = (const float*)d_in[1];   // [4,1,1,1]
    const float* wm = (const float*)d_in[2];   // [1,1,40,40]
    const float* wr = (const float*)d_in[3];   // [4,8,8,3,3]
    const float* wi = (const float*)d_in[4];   // [4,8,8,3,3]

    float* out = (float*)d_out;
    const size_t predSz = (size_t)NB * NC * HH * WW * 2;  // 26,214,400
    const size_t adjSz  = (size_t)NB * HH * WW;           //  1,638,400
    float* outAdj = out + predSz;
    float* outBin = out + predSz + adjSz;

    float* ping   = (float*)d_ws;        // intermediate pred buffer
    float* wsMask = ping + predSz;       // [4][40][40] binary mask

    mask_kernel<<<NB, 256, 0, stream>>>(wm, m1, outAdj, outBin, wsMask);

    dim3 grid(WW / TW, HH / TH, NB);     // 10 x 40 x 4
    const int WS = 8 * 8 * 9;            // weights per stack
    // scan over 4 stacks, ping-pong ws <-> d_out pred region
    spirit_step<<<grid, 256, 0, stream>>>(x,    x, wsMask, wr + 0 * WS, wi + 0 * WS, ping, 1);
    spirit_step<<<grid, 256, 0, stream>>>(ping, x, wsMask, wr + 1 * WS, wi + 1 * WS, out,  0);
    spirit_step<<<grid, 256, 0, stream>>>(out,  x, wsMask, wr + 2 * WS, wi + 2 * WS, ping, 0);
    spirit_step<<<grid, 256, 0, stream>>>(ping, x, wsMask, wr + 3 * WS, wi + 3 * WS, out,  0);
}
